// Bottleneck_42408507081100
// MI455X (gfx1250) — 8-bit matrix operand order repaired and checked by reading, compiled, not run on hardware
//
#include <hip/hip_runtime.h>


typedef __attribute__((ext_vector_type(8)))  int   v8i;
typedef __attribute__((ext_vector_type(16))) int   v16i;
typedef __attribute__((ext_vector_type(8)))  float v8f;

#define HWD 196
#define MTILES 98
#define FP8_ONE 0x38

__device__ __forceinline__ v16i cat(v8i a, v8i b) {
  return __builtin_shufflevector(a, b, 0, 1, 2, 3, 4, 5, 6, 7, 8, 9, 10, 11, 12, 13, 14, 15);
}

__device__ __forceinline__ size_t a_off(int xb, int m, int k, int Mt, int Kt) {
  int kl = k & 63;
  int lane = (m & 15) + (((kl >> 3) & 1) << 4);
  int byteIdx = ((kl >> 4) << 3) + (kl & 7);
  return ((((size_t)(xb * Mt + (m >> 4)) * (size_t)Kt + (size_t)(k >> 6)) << 10)
          + (size_t)(lane * 32 + byteIdx));
}
__device__ __forceinline__ size_t b_off(int wb, int n, int k, int Nt, int Kt) {
  int kl = k & 63;
  int lane = (n & 15) + (((kl >> 3) & 1) << 4);
  int byteIdx = ((kl >> 4) << 3) + (kl & 7);
  return ((((size_t)(wb * Nt + (n >> 4)) * (size_t)Kt + (size_t)(k >> 6)) << 10)
          + (size_t)(lane * 32 + byteIdx));
}

__global__ void reduce_max_kernel(const float* __restrict__ p, long n, int useAbs,
                                  unsigned* __restrict__ slot) {
  __shared__ float sm[256];
  float m = 0.0f;
  for (long i = (long)blockIdx.x * blockDim.x + threadIdx.x; i < n;
       i += (long)gridDim.x * blockDim.x) {
    float v = p[i];
    if (useAbs) v = fabsf(v);
    m = fmaxf(m, v);
  }
  sm[threadIdx.x] = m;
  __syncthreads();
  for (int s = 128; s > 0; s >>= 1) {
    if ((int)threadIdx.x < s) sm[threadIdx.x] = fmaxf(sm[threadIdx.x], sm[threadIdx.x + s]);
    __syncthreads();
  }
  if (threadIdx.x == 0) atomicMax(slot, __float_as_uint(sm[0]));
}

__global__ void pack_x1_kernel(const float* __restrict__ x, unsigned char* __restrict__ A1,
                               const unsigned* __restrict__ slots) {
  int idx = blockIdx.x * blockDim.x + threadIdx.x;
  if (idx >= 1024 * 1568) return;
  int c = idx / 1568, m = idx - c * 1568;
  int b = m / HWD, hw = m - b * HWD;
  float s = __uint_as_float(slots[0]) * (1.0f / 15.0f) + 1e-8f;
  float v = x[((size_t)b * 1024 + c) * HWD + hw];
  int q = __float2int_rn(fminf(fmaxf(v / s, 0.0f), 15.0f));
#pragma unroll
  for (int xb = 0; xb < 4; ++xb)
    A1[a_off(xb, m, c, MTILES, 16)] = (unsigned char)(FP8_ONE * ((q >> xb) & 1));
}

__global__ void pack_w_kernel(const float* __restrict__ w, unsigned char* __restrict__ Bb,
                              const unsigned* __restrict__ slots, int slot,
                              int O, int C, int taps, int is3x3, int Nt, int Kt) {
  int idx = blockIdx.x * blockDim.x + threadIdx.x;
  if (idx >= O * C * taps) return;
  int t = idx % taps;
  int rest = idx / taps;
  int c = rest % C, o = rest / C;
  float s = __uint_as_float(slots[slot]) * (1.0f / 7.0f) + 1e-8f;
  float v = w[idx];
  int q = __float2int_rn(fminf(fmaxf(v / s, -8.0f), 7.0f));
  int wi = q & 15;
  int k = is3x3 ? ((c / 28) * 256 + (c % 28) * 9 + t) : c;
#pragma unroll
  for (int wb = 0; wb < 4; ++wb)
    Bb[b_off(wb, o, k, Nt, Kt)] = (unsigned char)(FP8_ONE * ((wi >> wb) & 1));
}

__global__ void pack_x2_kernel(const float* __restrict__ Y1, unsigned char* __restrict__ A2,
                               const unsigned* __restrict__ slots) {
  int idx = blockIdx.x * blockDim.x + threadIdx.x;
  if (idx >= 256 * 1568) return;
  int c = idx / 1568, m = idx - c * 1568;
  int b = m / HWD, hw = m - b * HWD;
  int h = hw / 14, w = hw - h * 14;
  float s = __uint_as_float(slots[4]) * (1.0f / 15.0f) + 1e-8f;
  int kbase = (c / 28) * 256 + (c % 28) * 9;
#pragma unroll
  for (int dy = 0; dy < 3; ++dy)
#pragma unroll
    for (int dx = 0; dx < 3; ++dx) {
      int hs = h + dy - 1, ws = w + dx - 1;
      if ((unsigned)hs >= 14u || (unsigned)ws >= 14u) continue;
      float v = Y1[(size_t)(b * HWD + hs * 14 + ws) * 256 + c];
      int q = __float2int_rn(fminf(fmaxf(v / s, 0.0f), 15.0f));
      int k = kbase + dy * 3 + dx;
#pragma unroll
      for (int xb = 0; xb < 4; ++xb)
        A2[a_off(xb, m, k, MTILES, 38)] = (unsigned char)(FP8_ONE * ((q >> xb) & 1));
    }
}

__global__ void pack_x3_kernel(const float* __restrict__ Y2, unsigned char* __restrict__ A3,
                               const unsigned* __restrict__ slots) {
  int idx = blockIdx.x * blockDim.x + threadIdx.x;
  if (idx >= 256 * 1568) return;
  int c = idx / 1568, m = idx - c * 1568;
  float s = __uint_as_float(slots[5]) * (1.0f / 15.0f) + 1e-8f;
  float v = Y2[(size_t)m * 256 + c];
  int q = __float2int_rn(fminf(fmaxf(v / s, 0.0f), 15.0f));
#pragma unroll
  for (int xb = 0; xb < 4; ++xb)
    A3[a_off(xb, m, c, MTILES, 4)] = (unsigned char)(FP8_ONE * ((q >> xb) & 1));
}

template <bool FINAL>
__global__ __launch_bounds__(128) void asim_gemm(
    const unsigned char* __restrict__ A, const unsigned char* __restrict__ B, int Ntiles,
    int KtA  , int Fc  , int Tail  ,
    const unsigned* __restrict__ slots, int sx_slot, int sw_slot,
    const float* __restrict__ gam, const float* __restrict__ bet, const float* __restrict__ mu,
    const float* __restrict__ var, float* __restrict__ Y, unsigned* __restrict__ omax,
    const float* __restrict__ resid, float* __restrict__ OUT) {
  const int lane = threadIdx.x & 31;
  const int wid = threadIdx.x >> 5;
  const int ngroups = Ntiles >> 2;
  const int mtile = blockIdx.x / ngroups;
  const int ntile = (blockIdx.x % ngroups) * 4 + wid;
  const size_t laneoff = (size_t)lane * 32u;
  const size_t axstride = ((size_t)MTILES * (size_t)KtA) << 10;
  const size_t bwstride = ((size_t)Ntiles * (size_t)KtA) << 10;

  const unsigned char* Abase = A + (((size_t)mtile * (size_t)KtA) << 10) + laneoff;
  const unsigned char* Bbase = B + (((size_t)ntile * (size_t)KtA) << 10) + laneoff;

  float Sf[8] = {0.f, 0.f, 0.f, 0.f, 0.f, 0.f, 0.f, 0.f};

  int kt = 0;
  for (int chk = 0; chk < Fc; ++chk, kt += 4) {
    __builtin_prefetch(Abase + ((size_t)(kt + 4) << 10), 0, 1);
    __builtin_prefetch(Bbase + ((size_t)(kt + 4) << 10), 0, 1);
    v8i Areg[4][4];
#pragma unroll
    for (int xb = 0; xb < 4; ++xb)
#pragma unroll
      for (int i = 0; i < 4; ++i)
        Areg[xb][i] = *(const v8i*)(Abase + xb * axstride + ((size_t)(kt + i) << 10));
#pragma unroll
    for (int wb = 0; wb < 4; ++wb) {
      v8i Breg[4];
#pragma unroll
      for (int i = 0; i < 4; ++i)
        Breg[i] = *(const v8i*)(Bbase + wb * bwstride + ((size_t)(kt + i) << 10));
      const v16i B01 = cat(Breg[0], Breg[1]);
      const v16i B23 = cat(Breg[2], Breg[3]);
#pragma unroll
      for (int xb = 0; xb < 4; ++xb) {
        v8f cacc = {0.f, 0.f, 0.f, 0.f, 0.f, 0.f, 0.f, 0.f};
        cacc = __builtin_amdgcn_wmma_f32_16x16x128_fp8_fp8(cat(Areg[xb][0], Areg[xb][1]), B01,
                                                           (short)0, cacc, false, false);
        cacc = __builtin_amdgcn_wmma_f32_16x16x128_fp8_fp8(cat(Areg[xb][2], Areg[xb][3]), B23,
                                                           (short)0, cacc, false, false);
        const float cf = (float)((1 << xb) * ((wb == 3) ? -8 : (1 << wb)));
#pragma unroll
        for (int e = 0; e < 8; ++e) {
          float p = cacc[e];
          float r = (p >= 129.0f) ? (p - 1.0f) : p;
          Sf[e] = fmaf(cf, r, Sf[e]);
        }
      }
    }
  }
  if (Tail) {
    v16i Ax[4];
#pragma unroll
    for (int xb = 0; xb < 4; ++xb)
      Ax[xb] = cat(*(const v8i*)(Abase + xb * axstride + ((size_t)kt << 10)),
                   *(const v8i*)(Abase + xb * axstride + ((size_t)(kt + 1) << 10)));
#pragma unroll
    for (int wb = 0; wb < 4; ++wb) {
      const v16i Bx = cat(*(const v8i*)(Bbase + wb * bwstride + ((size_t)kt << 10)),
                          *(const v8i*)(Bbase + wb * bwstride + ((size_t)(kt + 1) << 10)));
#pragma unroll
      for (int xb = 0; xb < 4; ++xb) {
        v8f cacc = {0.f, 0.f, 0.f, 0.f, 0.f, 0.f, 0.f, 0.f};
        cacc = __builtin_amdgcn_wmma_f32_16x16x128_fp8_fp8(Ax[xb], Bx, (short)0, cacc,
                                                           false, false);
        const float cf = (float)((1 << xb) * ((wb == 3) ? -8 : (1 << wb)));
#pragma unroll
        for (int e = 0; e < 8; ++e) {
          float p = cacc[e];
          float r = (p >= 129.0f) ? (p - 1.0f) : p;
          Sf[e] = fmaf(cf, r, Sf[e]);
        }
      }
    }
  }

  const float sx = __uint_as_float(slots[sx_slot]) * (1.0f / 15.0f) + 1e-8f;
  const float sw = __uint_as_float(slots[sw_slot]) * (1.0f / 7.0f) + 1e-8f;
  const float scale = sx * sw * (256.0f / 255.0f);
  const int col = ntile * 16 + (lane & 15);
  const int rbase = mtile * 16 + ((lane >> 4) << 3);
  const float inv = gam[col] * rsqrtf(var[col] + 1e-5f);
  const float bias = bet[col] - mu[col] * inv;

  if (!FINAL) {
    const int O = Ntiles * 16;
    float lmax = 0.0f;
#pragma unroll
    for (int e = 0; e < 8; ++e) {
      int m = rbase + e;
      float z = fmaxf(Sf[e] * scale * inv + bias, 0.0f);
      Y[(size_t)m * O + col] = z;
      lmax = fmaxf(lmax, z);
    }
    atomicMax(omax, __float_as_uint(lmax));
  } else {
#pragma unroll
    for (int e = 0; e < 8; ++e) {
      int m = rbase + e;
      int b = m / HWD, hw = m - b * HWD;
      size_t oidx = ((size_t)b * 1024 + col) * HWD + hw;
      OUT[oidx] = fmaxf(Sf[e] * scale * inv + bias + resid[oidx], 0.0f);
    }
  }
}

extern "C" void kernel_launch(void* const* d_in, const int* in_sizes, int n_in, void* d_out,
                              int out_size, void* d_ws, size_t ws_size, hipStream_t stream) {
  (void)in_sizes; (void)n_in; (void)out_size; (void)ws_size;
  const float* x  = (const float*)d_in[0];
  const float* w1 = (const float*)d_in[1];
  const float* g1 = (const float*)d_in[2];
  const float* b1 = (const float*)d_in[3];
  const float* m1 = (const float*)d_in[4];
  const float* v1 = (const float*)d_in[5];
  const float* w2 = (const float*)d_in[6];
  const float* g2 = (const float*)d_in[7];
  const float* b2 = (const float*)d_in[8];
  const float* m2 = (const float*)d_in[9];
  const float* v2 = (const float*)d_in[10];
  const float* w3 = (const float*)d_in[11];
  const float* g3 = (const float*)d_in[12];
  const float* b3 = (const float*)d_in[13];
  const float* m3 = (const float*)d_in[14];
  const float* v3 = (const float*)d_in[15];
  float* out = (float*)d_out;

  char* ws = (char*)d_ws;
  const size_t SZ_A1 = 4ull * 98 * 16 * 1024, SZ_B1 = 4ull * 16 * 16 * 1024;
  const size_t SZ_Y = 1568ull * 256 * 4;
  const size_t SZ_A2 = 4ull * 98 * 38 * 1024, SZ_B2 = 4ull * 16 * 38 * 1024;
  const size_t SZ_A3 = 4ull * 98 * 4 * 1024, SZ_B3 = 4ull * 64 * 4 * 1024;
  size_t off = 256;
  unsigned* slots = (unsigned*)ws;
  unsigned char* A1 = (unsigned char*)(ws + off); off += SZ_A1;
  unsigned char* B1 = (unsigned char*)(ws + off); off += SZ_B1;
  float* Y1 = (float*)(ws + off); off += SZ_Y;
  float* Y2 = (float*)(ws + off); off += SZ_Y;
  unsigned char* A3 = (unsigned char*)(ws + off); off += SZ_A3;
  unsigned char* B3 = (unsigned char*)(ws + off); off += SZ_B3;
  unsigned char* A2 = (unsigned char*)(ws + off); off += SZ_A2;
  unsigned char* B2 = (unsigned char*)(ws + off); off += SZ_B2;

  hipMemsetAsync(slots, 0, 32, stream);
  hipMemsetAsync(A2, 0, SZ_A2 + SZ_B2, stream);

  reduce_max_kernel<<<512, 256, 0, stream>>>(x, 1605632L, 0, slots + 0);
  reduce_max_kernel<<<256, 256, 0, stream>>>(w1, 262144L, 1, slots + 1);
  reduce_max_kernel<<<256, 256, 0, stream>>>(w2, 589824L, 1, slots + 2);
  reduce_max_kernel<<<256, 256, 0, stream>>>(w3, 1048576L, 1, slots + 3);

  pack_w_kernel<<<262144 / 256, 256, 0, stream>>>(w1, B1, slots, 1, 256, 1024, 1, 0, 16, 16);
  pack_w_kernel<<<589824 / 256, 256, 0, stream>>>(w2, B2, slots, 2, 256, 256, 9, 1, 16, 38);
  pack_w_kernel<<<262144 / 256, 256, 0, stream>>>(w3, B3, slots, 3, 1024, 256, 1, 0, 64, 4);
  pack_x1_kernel<<<1605632 / 256, 256, 0, stream>>>(x, A1, slots);

  asim_gemm<false><<<98 * 4, 128, 0, stream>>>(A1, B1, 16, 16, 4, 0, slots, 0, 1, g1, b1, m1,
                                               v1, Y1, slots + 4, nullptr, nullptr);
  pack_x2_kernel<<<401408 / 256, 256, 0, stream>>>(Y1, A2, slots);
  asim_gemm<false><<<98 * 4, 128, 0, stream>>>(A2, B2, 16, 38, 9, 1, slots, 4, 2, g2, b2, m2,
                                               v2, Y2, slots + 5, nullptr, nullptr);
  pack_x3_kernel<<<401408 / 256, 256, 0, stream>>>(Y2, A3, slots);
  asim_gemm<true><<<98 * 16, 128, 0, stream>>>(A3, B3, 64, 4, 1, 0, slots, 5, 3, g3, b3, m3,
                                               v3, nullptr, nullptr, x, out);
}
